// QuantizedMLP_5291399709433
// MI455X (gfx1250) — matrix operand order repaired and checked by reading, not compiled, not run on hardware
//
#include <hip/hip_runtime.h>
#include <stdint.h>

// ---------------------------------------------------------------------------
// Types for CDNA5 WMMA (wave32)
// ---------------------------------------------------------------------------
typedef __attribute__((ext_vector_type(16))) __bf16 v16bf;
typedef __attribute__((ext_vector_type(8)))  __bf16 v8bf;
typedef __attribute__((ext_vector_type(8)))  float  v8f;

// ---------------------------------------------------------------------------
// Helpers
// ---------------------------------------------------------------------------
__device__ __forceinline__ unsigned short f2bf(float f) {
  unsigned int u = __float_as_uint(f);
  unsigned int r = u + 0x7FFFu + ((u >> 16) & 1u);   // round-to-nearest-even
  return (unsigned short)(r >> 16);
}

// Per-lane async copy of 16B global -> LDS (gfx1250, tracked by ASYNCcnt).
__device__ __forceinline__ void async_copy_b128(unsigned lds_off, const void* gptr) {
  asm volatile("global_load_async_to_lds_b128 %0, %1, off"
               :: "v"(lds_off), "v"(gptr) : "memory");
}
__device__ __forceinline__ unsigned lds_offset(const void* p) {
  // addrspace(3) flat address: low 32 bits are the LDS byte offset
  return (unsigned)(uintptr_t)p;
}

// Babai nearest-plane rounding on the E8 generator.
// z_j = round(sum_k x_k * GINV[k][j]) with closed-form GINV columns,
// y = z @ G reconstructed directly.
__device__ __forceinline__ void babai8(const float x[8], float y[8]) {
  float s6 = x[6];
  float s5 = s6 + x[5];
  float s4 = s5 + x[4];
  float s3 = s4 + x[3];
  float s2 = s3 + x[2];
  float s1 = s2 + x[1];
  float s0 = s1 + x[0];
  float x7 = x[7];
  float z0 = rintf(0.5f * s0 - 3.5f * x7);
  float z1 = rintf(s1 - 6.0f * x7);
  float z2 = rintf(s2 - 5.0f * x7);
  float z3 = rintf(s3 - 4.0f * x7);
  float z4 = rintf(s4 - 3.0f * x7);
  float z5 = rintf(s5 - 2.0f * x7);
  float z6 = rintf(s6 - x7);
  float z7 = rintf(2.0f * x7);
  float h7 = 0.5f * z7;
  y[0] = 2.0f * z0 - z1 + h7;
  y[1] = z1 - z2 + h7;
  y[2] = z2 - z3 + h7;
  y[3] = z3 - z4 + h7;
  y[4] = z4 - z5 + h7;
  y[5] = z5 - z6 + h7;
  y[6] = z6 + h7;
  y[7] = h7;
}

// ---------------------------------------------------------------------------
// Kernel: zero the stats accumulators (sum_h, absdev_h, sum_y, absdev_y)
// ---------------------------------------------------------------------------
__global__ void k_init_stats(float* __restrict__ stats) {
  if (threadIdx.x < 8) stats[threadIdx.x] = 0.0f;
}

// ---------------------------------------------------------------------------
// Kernel: per-row std (ddof=1) -> beta_r = Ymax/(C_B*sigma) = 0.45/sigma
// One block per row.
// ---------------------------------------------------------------------------
__global__ __launch_bounds__(256)
void k_rowstd_beta(const float* __restrict__ W, int cols, float* __restrict__ beta) {
  const int row = blockIdx.x;
  const float* p = W + (size_t)row * cols;
  float s = 0.f, s2 = 0.f;
  for (int c = threadIdx.x; c < cols; c += blockDim.x) {
    float v = p[c];
    s  += v;
    s2 += v * v;
  }
  __shared__ float rs[256];
  __shared__ float rq[256];
  rs[threadIdx.x] = s;
  rq[threadIdx.x] = s2;
  __syncthreads();
  for (int off = 128; off > 0; off >>= 1) {
    if ((int)threadIdx.x < off) {
      rs[threadIdx.x] += rs[threadIdx.x + off];
      rq[threadIdx.x] += rq[threadIdx.x + off];
    }
    __syncthreads();
  }
  if (threadIdx.x == 0) {
    float n   = (float)cols;
    float var = (rq[0] - rs[0] * rs[0] / n) / (n - 1.0f);
    var = fmaxf(var, 0.0f);
    float sigma = sqrtf(var) + 1e-8f;
    beta[row] = 0.45f / sigma;   // DELTA0*(q^M-1)/2 / (C_B*sigma) with q=4,M=2
  }
}

// ---------------------------------------------------------------------------
// Kernel: E8 nested-lattice quantize weights (q=4, M=2 => q^M = 16),
// clip to [-2,2], store bf16.  One thread per 8-element block.
// ---------------------------------------------------------------------------
__global__ __launch_bounds__(256)
void k_quant_e8(const float* __restrict__ W, const float* __restrict__ beta,
                int cols, unsigned short* __restrict__ Wq, long nblk) {
  long idx = (long)blockIdx.x * blockDim.x + threadIdx.x;
  if (idx >= nblk) return;
  const int cb  = cols >> 3;
  const int row = (int)(idx / cb);
  const float br = beta[row];
  const float* p = W + idx * 8;

  float v[8], QL[8], g[8], t[8];
#pragma unroll
  for (int i = 0; i < 8; i++) v[i] = br * p[i];
  babai8(v, QL);
  // g = babai(x)/q ; g = babai(g)/q ; r = babai(g)
#pragma unroll
  for (int i = 0; i < 8; i++) g[i] = QL[i] * 0.25f;
  babai8(g, t);
#pragma unroll
  for (int i = 0; i < 8; i++) g[i] = t[i] * 0.25f;
  babai8(g, t);  // t = r

  const float invb = 1.0f / br;
  unsigned short* o = Wq + idx * 8;
#pragma unroll
  for (int i = 0; i < 8; i++) {
    float wq = QL[i] - 16.0f * t[i];
    float wo = (wq * invb) / (1.0f + 1e-8f);
    wo = fminf(fmaxf(wo, -2.0f), 2.0f);
    o[i] = f2bf(wo);
  }
}

// ---------------------------------------------------------------------------
// Kernel: f32 -> bf16 conversion (grid-stride)
// ---------------------------------------------------------------------------
__global__ __launch_bounds__(256)
void k_f32_to_bf16(const float* __restrict__ in, unsigned short* __restrict__ out, long n) {
  long i = (long)blockIdx.x * blockDim.x + threadIdx.x;
  const long stride = (long)gridDim.x * blockDim.x;
  for (; i < n; i += stride) out[i] = f2bf(in[i]);
}

// ---------------------------------------------------------------------------
// Kernel: bf16 WMMA GEMM  C[m,n] = sum_k A[m,k]*Bt[n,k] + bias[n]
// Block: 256 threads = 8 wave32 waves, block tile 64(M) x 128(N), BK=32.
// A/B tiles double-buffered in LDS via global_load_async_to_lds_b128
// (ASYNCcnt), fragments read from LDS (ds_load_b128), 2x2
// v_wmma_f32_16x16x32_bf16 accumulators per wave (32x32 output/wave).
// Also accumulates the per-tensor sum (for ACIQ mean) into *sumAcc.
// Fragment layouts per CDNA5 ISA 7.12.2 (16-bit A 16x32 / B 32x16).
// ---------------------------------------------------------------------------
__global__ __launch_bounds__(256)
void k_gemm_wmma(const unsigned short* __restrict__ A, const unsigned short* __restrict__ Bt,
                 const float* __restrict__ bias, float* __restrict__ C,
                 float* __restrict__ sumAcc, int Mdim, int Ndim, int Kdim) {
  __shared__ __align__(16) unsigned short sA[2][64 * 32];    // 2 x 4 KB
  __shared__ __align__(16) unsigned short sB[2][128 * 32];   // 2 x 8 KB

  const int tid    = threadIdx.x;
  const int lane   = tid & 31;
  const int wave   = tid >> 5;
  const int waveM  = wave >> 2;            // 0..1
  const int waveN  = wave & 3;             // 0..3
  const int bm     = blockIdx.y * 64;
  const int bn     = blockIdx.x * 128;
  const int laneHi = lane >> 4;            // 0 or 1
  const int lmod   = lane & 15;

  // Staging assignment: 16B per thread for A (4 KB), 2x16B for B (8 KB).
  const int arow = tid >> 2;               // 0..63
  const int acg  = (tid & 3) * 8;          // K element sub-offset

  auto issue_tile = [&](int buf, int k0) {
    {
      const unsigned short* gp = A + (size_t)(bm + arow) * Kdim + (size_t)k0 + acg;
      async_copy_b128(lds_offset(&sA[buf][arow * 32 + acg]), gp);
    }
#pragma unroll
    for (int j = 0; j < 2; j++) {
      int idx = tid + j * 256;
      int row = idx >> 2;                  // 0..127
      int cg  = (idx & 3) * 8;
      const unsigned short* gp = Bt + (size_t)(bn + row) * Kdim + (size_t)k0 + cg;
      async_copy_b128(lds_offset(&sB[buf][row * 32 + cg]), gp);
    }
  };

  v8f acc[2][2];
#pragma unroll
  for (int i = 0; i < 2; i++)
#pragma unroll
    for (int j = 0; j < 2; j++)
#pragma unroll
      for (int r = 0; r < 8; r++) acc[i][j][r] = 0.0f;

  const int nk = Kdim >> 5;
  issue_tile(0, 0);

  for (int kt = 0; kt < nk; ++kt) {
    const int buf = kt & 1;
    if (kt + 1 < nk) {
      issue_tile(buf ^ 1, (kt + 1) << 5);
      // asyncs complete in order per wave: <=3 outstanding => tile kt landed
      asm volatile("s_wait_asynccnt 0x3" ::: "memory");
    } else {
      asm volatile("s_wait_asynccnt 0x0" ::: "memory");
    }
    __syncthreads();

    const unsigned short* a0 = sA[buf];
    const unsigned short* b0 = sB[buf];
    v16bf afr[2], bfr[2];
    // A fragments: 16x32, lanes 0-15 rows M, K = {0..7,16..23}; lanes 16-31 K = {8..15,24..31}
#pragma unroll
    for (int t = 0; t < 2; t++) {
      int row = waveM * 32 + t * 16 + lmod;
      v8bf lo = *reinterpret_cast<const v8bf*>(a0 + row * 32 + laneHi * 8);
      v8bf hi = *reinterpret_cast<const v8bf*>(a0 + row * 32 + laneHi * 8 + 16);
      afr[t] = __builtin_shufflevector(lo, hi, 0, 1, 2, 3, 4, 5, 6, 7,
                                               8, 9, 10, 11, 12, 13, 14, 15);
    }
    // B fragments: 32x16, lane = column N, lanes 0-15 hold K = {0..7,16..23}; lanes 16-31 K = {8..15,24..31}
#pragma unroll
    for (int t = 0; t < 2; t++) {
      int row = waveN * 32 + t * 16 + lmod;
      v8bf lo = *reinterpret_cast<const v8bf*>(b0 + row * 32 + laneHi * 8);
      v8bf hi = *reinterpret_cast<const v8bf*>(b0 + row * 32 + laneHi * 8 + 16);
      bfr[t] = __builtin_shufflevector(lo, hi, 0, 1, 2, 3, 4, 5, 6, 7,
                                               8, 9, 10, 11, 12, 13, 14, 15);
    }
#pragma unroll
    for (int i = 0; i < 2; i++)
#pragma unroll
      for (int j = 0; j < 2; j++)
        acc[i][j] = __builtin_amdgcn_wmma_f32_16x16x32_bf16(
            /*neg_a=*/false, afr[i], /*neg_b=*/false, bfr[j],
            /*c_mod=*/(short)0, acc[i][j], /*reuse_a=*/false, /*reuse_b=*/false);

    // Protect buf (it is the async target in the next iteration)
    __syncthreads();
  }

  // Epilogue: bias add, store, fused per-tensor sum.
  // C/D layout: VGPR r, lanes 0-15 -> M=r, N=lane; lanes 16-31 -> M=8+r, N=lane-16.
  float lsum = 0.0f;
#pragma unroll
  for (int i = 0; i < 2; i++) {
#pragma unroll
    for (int j = 0; j < 2; j++) {
      const int nn    = bn + waveN * 32 + j * 16 + lmod;
      const float bv  = bias[nn];
      const int mbase = bm + waveM * 32 + i * 16 + laneHi * 8;
#pragma unroll
      for (int r = 0; r < 8; r++) {
        float v = acc[i][j][r] + bv;
        C[(size_t)(mbase + r) * Ndim + nn] = v;
        lsum += v;
      }
    }
  }
  __shared__ float red[256];
  red[threadIdx.x] = lsum;
  __syncthreads();
  for (int off = 128; off > 0; off >>= 1) {
    if ((int)threadIdx.x < off) red[threadIdx.x] += red[threadIdx.x + off];
    __syncthreads();
  }
  if (threadIdx.x == 0) atomicAdd(sumAcc, red[0]);
}

// ---------------------------------------------------------------------------
// Kernel: sum |v - mu| (mu = *sumSlot / n) -> atomicAdd into *dst
// ---------------------------------------------------------------------------
__global__ __launch_bounds__(256)
void k_absdev(const float* __restrict__ v, long n,
              const float* __restrict__ sumSlot, float* __restrict__ dst) {
  const float mu = (*sumSlot) / (float)n;
  float s = 0.0f;
  long i = (long)blockIdx.x * blockDim.x + threadIdx.x;
  const long stride = (long)gridDim.x * blockDim.x;
  for (; i < n; i += stride) s += fabsf(v[i] - mu);
  __shared__ float red[256];
  red[threadIdx.x] = s;
  __syncthreads();
  for (int off = 128; off > 0; off >>= 1) {
    if ((int)threadIdx.x < off) red[threadIdx.x] += red[threadIdx.x + off];
    __syncthreads();
  }
  if (threadIdx.x == 0) atomicAdd(dst, red[0]);
}

// ---------------------------------------------------------------------------
// Kernel: ACIQ fake-quant + ReLU, f32 in -> bf16 out (for hidden activations)
// ---------------------------------------------------------------------------
__global__ __launch_bounds__(256)
void k_aciq_relu_bf16(const float* __restrict__ in, unsigned short* __restrict__ out,
                      long n, const float* __restrict__ absSlot) {
  const float b     = (*absSlot) / (float)n;
  const float alpha = 9.89f * b;
  const float scale = alpha / 127.0f;
  const float inv   = 127.0f / alpha;
  long i = (long)blockIdx.x * blockDim.x + threadIdx.x;
  const long stride = (long)gridDim.x * blockDim.x;
  for (; i < n; i += stride) {
    float v  = in[i];
    float xc = fminf(fmaxf(v, -alpha), alpha);
    float xr = rintf(xc * inv) * scale;
    out[i] = f2bf(fmaxf(xr, 0.0f));
  }
}

// ---------------------------------------------------------------------------
// Kernel: ACIQ fake-quant in-place on f32 (final output)
// ---------------------------------------------------------------------------
__global__ __launch_bounds__(256)
void k_aciq_inplace(float* __restrict__ io, long n, const float* __restrict__ absSlot) {
  const float b     = (*absSlot) / (float)n;
  const float alpha = 9.89f * b;
  const float scale = alpha / 127.0f;
  const float inv   = 127.0f / alpha;
  long i = (long)blockIdx.x * blockDim.x + threadIdx.x;
  const long stride = (long)gridDim.x * blockDim.x;
  for (; i < n; i += stride) {
    float v  = io[i];
    float xc = fminf(fmaxf(v, -alpha), alpha);
    io[i] = rintf(xc * inv) * scale;
  }
}

// ---------------------------------------------------------------------------
// Host launcher
// ---------------------------------------------------------------------------
extern "C" void kernel_launch(void* const* d_in, const int* in_sizes, int n_in,
                              void* d_out, int out_size, void* d_ws, size_t ws_size,
                              hipStream_t stream) {
  (void)in_sizes; (void)n_in; (void)out_size; (void)ws_size;
  const float* x  = (const float*)d_in[0];   // [8192,1024]
  const float* W1 = (const float*)d_in[1];   // [4096,1024]
  const float* b1 = (const float*)d_in[2];   // [4096]
  const float* W2 = (const float*)d_in[3];   // [1024,4096]
  const float* b2 = (const float*)d_in[4];   // [1024]
  // d_in[5]=q(4), d_in[6]=M(2): constants baked in (q^M = 16).
  float* out = (float*)d_out;                // [8192,1024]

  const int Nrow = 8192, D = 1024, H = 4096;

  char* ws = (char*)d_ws;
  float* stats          = (float*)ws;                         // [0]=sum_h [1]=ad_h [2]=sum_y [3]=ad_y
  float* beta1          = (float*)(ws + 256);                 // 4096 f32
  float* beta2          = (float*)(ws + 256 + 4096 * 4);      // 1024 f32
  unsigned short* xb    = (unsigned short*)(ws + 32768);                                  // 16 MB
  unsigned short* w1q   = (unsigned short*)(ws + 32768 + (size_t)16 * 1024 * 1024);       //  8 MB
  unsigned short* w2q   = (unsigned short*)(ws + 32768 + (size_t)24 * 1024 * 1024);       //  8 MB
  float* hpre           = (float*)(ws + 32768 + (size_t)32 * 1024 * 1024);                // 128 MB
  unsigned short* hb    = (unsigned short*)(ws + 32768 + (size_t)160 * 1024 * 1024);      // 64 MB

  k_init_stats<<<1, 32, 0, stream>>>(stats);

  k_rowstd_beta<<<H, 256, 0, stream>>>(W1, D, beta1);
  k_rowstd_beta<<<D, 256, 0, stream>>>(W2, H, beta2);

  k_quant_e8<<<2048, 256, 0, stream>>>(W1, beta1, D, w1q, (long)H * D / 8);
  k_quant_e8<<<2048, 256, 0, stream>>>(W2, beta2, H, w2q, (long)D * H / 8);

  k_f32_to_bf16<<<4096, 256, 0, stream>>>(x, xb, (long)Nrow * D);

  dim3 g1(H / 128, Nrow / 64);   // 32 x 128
  k_gemm_wmma<<<g1, 256, 0, stream>>>(xb, w1q, b1, hpre, stats + 0, Nrow, H, D);

  k_absdev<<<4096, 256, 0, stream>>>(hpre, (long)Nrow * H, stats + 0, stats + 1);
  k_aciq_relu_bf16<<<4096, 256, 0, stream>>>(hpre, hb, (long)Nrow * H, stats + 1);

  dim3 g2(D / 128, Nrow / 64);   // 8 x 128
  k_gemm_wmma<<<g2, 256, 0, stream>>>(hb, w2q, b2, out, stats + 2, Nrow, D, H);

  k_absdev<<<2048, 256, 0, stream>>>(out, (long)Nrow * D, stats + 2, stats + 3);
  k_aciq_inplace<<<2048, 256, 0, stream>>>(out, (long)Nrow * D, stats + 3);
}
